// GNNAgentV2_84834194031328
// MI455X (gfx1250) — compile-verified, matrix operand order checked by reading, not run on hardware
//
#include <hip/hip_runtime.h>
#include <hip/hip_bf16.h>

typedef __attribute__((ext_vector_type(16))) _Float16 v16h;
typedef __attribute__((ext_vector_type(8)))  _Float16 v8h;
typedef __attribute__((ext_vector_type(8)))  float    v8f;

#define LEAKY  0.2f
#define LN_EPS 1e-5f
#define FDIM   128
#define HDIM   256
#define DDIM   64
#define HEADS  4
#define ACTN   32

// ---------- monotonic float <-> uint mapping for atomicMax on floats ----------
__device__ __forceinline__ unsigned f2ord(float f) {
    unsigned u = __float_as_uint(f);
    return (u & 0x80000000u) ? ~u : (u | 0x80000000u);
}
__device__ __forceinline__ float ord2f(unsigned u) {
    return __uint_as_float((u & 0x80000000u) ? (u & 0x7FFFFFFFu) : ~u);
}

// ---------- tiny utility kernels ----------
__global__ void fill_u32(unsigned* __restrict__ p, unsigned v, long long n) {
    long long i = (long long)blockIdx.x * blockDim.x + threadIdx.x;
    if (i < n) p[i] = v;
}

__global__ void cvt_f16(const float* __restrict__ x, _Float16* __restrict__ y, long long n) {
    long long i = (long long)blockIdx.x * blockDim.x + threadIdx.x;
    if (i < n) y[i] = (_Float16)x[i];
}

// W [K][Nout] f32 row-major  ->  Wt [Nout][K] f16 (K-contiguous per output column)
__global__ void wtrans(const float* __restrict__ W, _Float16* __restrict__ Wt, int K, int Nout) {
    long long i = (long long)blockIdx.x * blockDim.x + threadIdx.x;
    if (i < (long long)K * Nout) {
        int k = (int)(i / Nout), n = (int)(i % Nout);
        Wt[(size_t)n * K + k] = (_Float16)W[i];
    }
}

// ---------- WMMA GEMM:  out[M x NOUT] = A[M x K] * Bt[NOUT x K]^T  (+ fused epilogue) ----------
// EPI 0: raw f16 store.  EPI 1: +bias, ReLU, LayerNorm(gamma,beta) -> f16.  EPI 2: (+bias) -> f32.
// Block = WAVES*32 threads; each wave owns exactly TPW 16x16 column tiles (no divergent
// guards -> WMMA always executes with EXEC all-1s, straight-line unrolled K loop).
template<int K, int NOUT, int EPI>
__global__ __launch_bounds__(((NOUT / 16) >= 4 ? 4 : (NOUT / 16)) * 32) void gemm_wmma(
    const _Float16* __restrict__ A, const _Float16* __restrict__ Bt,
    const float* __restrict__ bias, const float* __restrict__ gamma, const float* __restrict__ beta,
    _Float16* __restrict__ out16, float* __restrict__ out32)
{
    constexpr int NTILES = NOUT / 16;
    constexpr int WAVES  = (NTILES >= 4) ? 4 : NTILES;
    constexpr int TPW    = NTILES / WAVES;   // exact for all instantiations
    constexpr int NTHR   = WAVES * 32;
    const int tid  = threadIdx.x;
    const int wave = tid >> 5;
    const int lane = tid & 31;
    const int half = lane >> 4;
    const int l16  = lane & 15;
    const int row0 = blockIdx.x * 16;

    v8f acc[TPW];
    #pragma unroll
    for (int t = 0; t < TPW; ++t)
        #pragma unroll
        for (int i = 0; i < 8; ++i) acc[t][i] = 0.0f;

    const _Float16* aRow = A + (size_t)(row0 + l16) * K;

    for (int k0 = 0; k0 < K; k0 += 32) {
        // A fragment: lane holds K in [half*8, half*8+8) and [16+half*8, 16+half*8+8)
        const _Float16* pa = aRow + k0 + half * 8;
        v8h alo = *(const v8h*)(pa);
        v8h ahi = *(const v8h*)(pa + 16);
        v16h af;
        #pragma unroll
        for (int i = 0; i < 8; ++i) { af[i] = alo[i]; af[8 + i] = ahi[i]; }
        if (k0 + 32 < K) __builtin_prefetch(aRow + k0 + 32, 0, 3);  // WGP-scope prefetch

        #pragma unroll
        for (int t = 0; t < TPW; ++t) {
            const int col = (wave * TPW + t) * 16 + l16;
            const _Float16* pb = Bt + (size_t)col * K + k0 + half * 8;
            v8h blo = *(const v8h*)(pb);
            v8h bhi = *(const v8h*)(pb + 16);
            v16h bf;
            #pragma unroll
            for (int i = 0; i < 8; ++i) { bf[i] = blo[i]; bf[8 + i] = bhi[i]; }
            acc[t] = __builtin_amdgcn_wmma_f32_16x16x32_f16(
                false, af, false, bf, (short)0, acc[t], false, false);
        }
    }

    // D layout: VGPR r, lane -> M = half*8 + r, N = l16
    if constexpr (EPI == 0) {
        #pragma unroll
        for (int t = 0; t < TPW; ++t) {
            const int col = (wave * TPW + t) * 16 + l16;
            #pragma unroll
            for (int r = 0; r < 8; ++r)
                out16[(size_t)(row0 + half * 8 + r) * NOUT + col] = (_Float16)acc[t][r];
        }
    } else if constexpr (EPI == 2) {
        #pragma unroll
        for (int t = 0; t < TPW; ++t) {
            const int col = (wave * TPW + t) * 16 + l16;
            const float bv = bias ? bias[col] : 0.0f;
            #pragma unroll
            for (int r = 0; r < 8; ++r)
                out32[(size_t)(row0 + half * 8 + r) * NOUT + col] = acc[t][r] + bv;
        }
    } else {  // EPI == 1 : bias + ReLU + LayerNorm -> f16
        __shared__ float tile_s[16][NOUT];
        __shared__ float mu_s[16], rs_s[16];
        #pragma unroll
        for (int t = 0; t < TPW; ++t) {
            const int col = (wave * TPW + t) * 16 + l16;
            const float bv = bias[col];
            #pragma unroll
            for (int r = 0; r < 8; ++r) {
                float v = acc[t][r] + bv;
                v = v > 0.0f ? v : 0.0f;
                tile_s[half * 8 + r][col] = v;
            }
        }
        __syncthreads();
        if (tid < 16) {
            float s = 0.f, s2 = 0.f;
            for (int j = 0; j < NOUT; ++j) { float v = tile_s[tid][j]; s += v; s2 += v * v; }
            float mu = s / (float)NOUT;
            float var = s2 / (float)NOUT - mu * mu;
            mu_s[tid] = mu;
            rs_s[tid] = rsqrtf(var + LN_EPS);
        }
        __syncthreads();
        for (int e = tid; e < 16 * NOUT; e += NTHR) {
            int rr = e / NOUT, cc = e % NOUT;
            float v = (tile_s[rr][cc] - mu_s[rr]) * rs_s[rr] * gamma[cc] + beta[cc];
            out16[(size_t)(row0 + rr) * NOUT + cc] = (_Float16)v;
        }
    }
}

// ---------- edge pass A: e[e,h] = att_h . leaky(xl[src] + xr[dst]) ; segment max ----------
// one wave per edge; 8 lanes per head, 8 channels per lane (f16 gathers hit L2: xl+xr ~51MB)
__global__ __launch_bounds__(256) void edge_scores(
    const _Float16* __restrict__ xl, const _Float16* __restrict__ xr,
    const int* __restrict__ src, const int* __restrict__ dst,
    const float* __restrict__ att, float* __restrict__ evals,
    unsigned* __restrict__ nodemax, int E)
{
    int wid = blockIdx.x * 8 + (threadIdx.x >> 5);
    if (wid >= E) return;
    int lane = threadIdx.x & 31;
    int s = src[wid], d = dst[wid];
    v8h vl = *(const v8h*)(xl + (size_t)s * (HEADS * DDIM) + lane * 8);
    v8h vr = *(const v8h*)(xr + (size_t)d * (HEADS * DDIM) + lane * 8);
    float part = 0.f;
    #pragma unroll
    for (int i = 0; i < 8; ++i) {
        float m = (float)vl[i] + (float)vr[i];
        m = (m > 0.f) ? m : LEAKY * m;
        part += m * att[lane * 8 + i];
    }
    part += __shfl_xor(part, 1, 32);
    part += __shfl_xor(part, 2, 32);
    part += __shfl_xor(part, 4, 32);
    if ((lane & 7) == 0) {
        int h = lane >> 3;
        evals[(size_t)wid * HEADS + h] = part;
        atomicMax(&nodemax[d * HEADS + h], f2ord(part));
    }
}

// ---------- edge pass B: p = exp(e - max[dst]); denom += p; acc[dst] += p * xl[src] ----------
__global__ __launch_bounds__(256) void edge_accum(
    const _Float16* __restrict__ xl, const int* __restrict__ src, const int* __restrict__ dst,
    const float* __restrict__ evals, const unsigned* __restrict__ nodemax,
    float* __restrict__ denom, float* __restrict__ acc, int E)
{
    int wid = blockIdx.x * 8 + (threadIdx.x >> 5);
    if (wid >= E) return;
    int lane = threadIdx.x & 31;
    int s = src[wid], d = dst[wid];
    int h = lane >> 3;
    float emax = ord2f(nodemax[d * HEADS + h]);
    float p = __expf(evals[(size_t)wid * HEADS + h] - emax);
    if ((lane & 7) == 0) atomicAdd(&denom[d * HEADS + h], p);
    v8h vl = *(const v8h*)(xl + (size_t)s * (HEADS * DDIM) + lane * 8);
    float* pa = acc + (size_t)d * (HEADS * DDIM) + lane * 8;
    #pragma unroll
    for (int i = 0; i < 8; ++i) atomicAdd(pa + i, p * (float)vl[i]);
}

// ---------- node combine: head-mean of acc/denom + residual + bias, ReLU, LayerNorm -> f16 ----------
__global__ __launch_bounds__(64) void node_combine(
    const float* __restrict__ acc, const float* __restrict__ denom,
    const float* __restrict__ res, const float* __restrict__ bias,
    const float* __restrict__ gamma, const float* __restrict__ beta,
    _Float16* __restrict__ hout)
{
    int n = blockIdx.x;
    int dcol = threadIdx.x;  // 0..63
    float s = 0.f;
    #pragma unroll
    for (int h = 0; h < HEADS; ++h)
        s += acc[(size_t)n * (HEADS * DDIM) + h * DDIM + dcol] / denom[n * HEADS + h];
    s = s * (1.0f / HEADS) + res[(size_t)n * DDIM + dcol] + bias[dcol];
    s = s > 0.f ? s : 0.f;
    __shared__ float sv[64], sq[64];
    sv[dcol] = s; sq[dcol] = s * s;
    __syncthreads();
    for (int st = 32; st > 0; st >>= 1) {
        if (dcol < st) { sv[dcol] += sv[dcol + st]; sq[dcol] += sq[dcol + st]; }
        __syncthreads();
    }
    float mu = sv[0] * (1.0f / DDIM);
    float var = sq[0] * (1.0f / DDIM) - mu * mu;
    float rs = rsqrtf(var + LN_EPS);
    hout[(size_t)n * DDIM + dcol] = (_Float16)((s - mu) * rs * gamma[dcol] + beta[dcol]);
}

// ---------- host orchestration ----------
extern "C" void kernel_launch(void* const* d_in, const int* in_sizes, int n_in,
                              void* d_out, int out_size, void* d_ws, size_t ws_size,
                              hipStream_t stream) {
    const float* inputs = (const float*)d_in[0];
    const int*   src    = (const int*)d_in[1];
    const int*   dst    = (const int*)d_in[2];
    const float *w0 = (const float*)d_in[3],  *b0 = (const float*)d_in[4];
    const float *g0 = (const float*)d_in[5],  *be0 = (const float*)d_in[6];
    const float *w1 = (const float*)d_in[7],  *b1 = (const float*)d_in[8];
    const float *g1 = (const float*)d_in[9],  *be1 = (const float*)d_in[10];
    const float *wl1 = (const float*)d_in[11], *wr1 = (const float*)d_in[12];
    const float *att1 = (const float*)d_in[13], *wres1 = (const float*)d_in[14];
    const float *bias1 = (const float*)d_in[15], *lg1 = (const float*)d_in[16], *lb1 = (const float*)d_in[17];
    const float *wl2 = (const float*)d_in[18], *wr2 = (const float*)d_in[19];
    const float *att2 = (const float*)d_in[20], *wres2 = (const float*)d_in[21];
    const float *bias2 = (const float*)d_in[22], *lg2 = (const float*)d_in[23], *lb2 = (const float*)d_in[24];
    const float *wa = (const float*)d_in[25], *ba = (const float*)d_in[26];

    const int N = in_sizes[0] / FDIM;
    const int E = in_sizes[1];

    char* ws = (char*)d_ws;
    size_t off = 0;
    auto alloc = [&](size_t bytes) -> char* {
        char* p = ws + off;
        off += (bytes + 255) & ~(size_t)255;
        return p;
    };

    _Float16* IN16 = (_Float16*)alloc((size_t)N * FDIM * 2);
    _Float16* X0   = (_Float16*)alloc((size_t)N * HDIM * 2);
    _Float16* X1   = (_Float16*)alloc((size_t)N * HDIM * 2);
    _Float16* XR   = (_Float16*)alloc((size_t)N * HDIM * 2);
    float*    EV   = (float*)alloc((size_t)E * HEADS * 4);
    unsigned* NMAX = (unsigned*)alloc((size_t)N * HEADS * 4);
    float*    DEN  = (float*)alloc((size_t)N * HEADS * 4);
    float*    ACC  = (float*)alloc((size_t)N * HDIM * 4);
    _Float16* H1   = (_Float16*)alloc((size_t)N * DDIM * 2);
    _Float16* H2   = (_Float16*)alloc((size_t)N * DDIM * 2);
    _Float16* W0T    = (_Float16*)alloc((size_t)FDIM * HDIM * 2);
    _Float16* W1T    = (_Float16*)alloc((size_t)HDIM * HDIM * 2);
    _Float16* WL1T   = (_Float16*)alloc((size_t)HDIM * HDIM * 2);
    _Float16* WR1T   = (_Float16*)alloc((size_t)HDIM * HDIM * 2);
    _Float16* WRES1T = (_Float16*)alloc((size_t)HDIM * DDIM * 2);
    _Float16* WL2T   = (_Float16*)alloc((size_t)DDIM * HDIM * 2);
    _Float16* WR2T   = (_Float16*)alloc((size_t)DDIM * HDIM * 2);
    _Float16* WRES2T = (_Float16*)alloc((size_t)DDIM * DDIM * 2);
    _Float16* WAT    = (_Float16*)alloc((size_t)DDIM * ACTN * 2);

    // Aliases: X0 is dead once X1 is built -> reuse as XL.
    //          IN16 is dead once X0 is built -> reuse as RES (N*64*4 == N*128*2 bytes).
    _Float16* XL  = X0;
    float*    RES = (float*)IN16;

    const long long nIn = (long long)N * FDIM;
    cvt_f16<<<(unsigned)((nIn + 255) / 256), 256, 0, stream>>>(inputs, IN16, nIn);

    auto launch_wtrans = [&](const float* W, _Float16* Wt, int K, int Nout) {
        long long n = (long long)K * Nout;
        wtrans<<<(unsigned)((n + 255) / 256), 256, 0, stream>>>(W, Wt, K, Nout);
    };
    launch_wtrans(w0, W0T, FDIM, HDIM);
    launch_wtrans(w1, W1T, HDIM, HDIM);
    launch_wtrans(wl1, WL1T, HDIM, HDIM);
    launch_wtrans(wr1, WR1T, HDIM, HDIM);
    launch_wtrans(wres1, WRES1T, HDIM, DDIM);
    launch_wtrans(wl2, WL2T, DDIM, HDIM);
    launch_wtrans(wr2, WR2T, DDIM, HDIM);
    launch_wtrans(wres2, WRES2T, DDIM, DDIM);
    launch_wtrans(wa, WAT, DDIM, ACTN);

    const unsigned gM = (unsigned)(N / 16);
    const unsigned gE = (unsigned)((E + 7) / 8);

    // base MLP: (Linear + ReLU + LN) x 2  — fused WMMA GEMMs
    gemm_wmma<FDIM, HDIM, 1><<<gM, 128, 0, stream>>>(IN16, W0T, b0, g0, be0, X0, nullptr);
    gemm_wmma<HDIM, HDIM, 1><<<gM, 128, 0, stream>>>(X0, W1T, b1, g1, be1, X1, nullptr);

    auto zero = [&](void* p, long long n32) {
        fill_u32<<<(unsigned)((n32 + 255) / 256), 256, 0, stream>>>((unsigned*)p, 0u, n32);
    };

    // ---- GATv2 layer 1 ----
    gemm_wmma<HDIM, HDIM, 0><<<gM, 128, 0, stream>>>(X1, WL1T, nullptr, nullptr, nullptr, XL, nullptr);
    gemm_wmma<HDIM, HDIM, 0><<<gM, 128, 0, stream>>>(X1, WR1T, nullptr, nullptr, nullptr, XR, nullptr);
    gemm_wmma<HDIM, DDIM, 2><<<gM, 128, 0, stream>>>(X1, WRES1T, nullptr, nullptr, nullptr, nullptr, RES);
    zero(NMAX, (long long)N * HEADS);
    zero(DEN,  (long long)N * HEADS);
    zero(ACC,  (long long)N * HDIM);
    edge_scores<<<gE, 256, 0, stream>>>(XL, XR, src, dst, att1, EV, NMAX, E);
    edge_accum<<<gE, 256, 0, stream>>>(XL, src, dst, EV, NMAX, DEN, ACC, E);
    node_combine<<<(unsigned)N, 64, 0, stream>>>(ACC, DEN, RES, bias1, lg1, lb1, H1);

    // ---- GATv2 layer 2 ----
    gemm_wmma<DDIM, HDIM, 0><<<gM, 128, 0, stream>>>(H1, WL2T, nullptr, nullptr, nullptr, XL, nullptr);
    gemm_wmma<DDIM, HDIM, 0><<<gM, 128, 0, stream>>>(H1, WR2T, nullptr, nullptr, nullptr, XR, nullptr);
    gemm_wmma<DDIM, DDIM, 2><<<gM, 128, 0, stream>>>(H1, WRES2T, nullptr, nullptr, nullptr, nullptr, RES);
    zero(NMAX, (long long)N * HEADS);
    zero(DEN,  (long long)N * HEADS);
    zero(ACC,  (long long)N * HDIM);
    edge_scores<<<gE, 256, 0, stream>>>(XL, XR, src, dst, att2, EV, NMAX, E);
    edge_accum<<<gE, 256, 0, stream>>>(XL, src, dst, EV, NMAX, DEN, ACC, E);
    node_combine<<<(unsigned)N, 64, 0, stream>>>(ACC, DEN, RES, bias2, lg2, lb2, H2);

    // action head: q = h2 @ wa + ba  -> f32 d_out (NTILES=2 -> 2 waves / 64 threads)
    gemm_wmma<DDIM, ACTN, 2><<<gM, 64, 0, stream>>>(H2, WAT, ba, nullptr, nullptr, nullptr, (float*)d_out);
}
